// LSTM_HEAD_44607530336395
// MI455X (gfx1250) — compile-verified, matrix operand order checked by reading, not run on hardware
//
#include <hip/hip_runtime.h>
#include <hip/hip_bf16.h>
#include <math.h>

// Problem dims (fixed by the reference)
#define B_   64
#define T_   512
#define C_   64
#define H_   512
#define KX   96          // x-part K padded: 64 ctx + 1 flow + 31 zeros -> 3 WMMA k-chunks
#define KTOT 608         // KX + H_
#define NKC  (KTOT/32)   // 19 WMMA k-chunks
#define NWG  64          // 32 gate-col groups x 2 batch halves
#define NTH  256         // 8 waves of 32

typedef __bf16 bf16;
typedef __attribute__((ext_vector_type(8)))  __bf16 v8bf;
typedef __attribute__((ext_vector_type(16))) __bf16 v16bf;
typedef __attribute__((ext_vector_type(8)))  float  v8f;

// Load one WMMA 16-bit operand fragment for this lane.
// Per CDNA5 ISA 7.12.2 (16-bit A 16x32): lane L holds row (L&15); elements are two
// contiguous 8-element K chunks at k0=(L>>4)*8 and k0+16. `p` points at
// row_base + k0 already; chunks are p[0..7] and p[16..23] (16B-aligned loads).
__device__ __forceinline__ v16bf ld_frag(const bf16* p) {
    v8bf lo = *reinterpret_cast<const v8bf*>(p);
    v8bf hi = *reinterpret_cast<const v8bf*>(p + 16);
    return __builtin_shufflevector(lo, hi, 0,1,2,3,4,5,6,7,8,9,10,11,12,13,14,15);
}

__device__ __forceinline__ float sigm_(float x) { return 1.0f / (1.0f + __expf(-x)); }

// ---------------- prep kernels ----------------

// Xpad[t][b][k]: k<64 -> x_context[b][t][k]; k==64 -> y_flow[b][t]; else 0. bf16.
__global__ void pack_x_kernel(const float* __restrict__ xc,
                              const float* __restrict__ yf,
                              bf16* __restrict__ xpad) {
    size_t idx = (size_t)blockIdx.x * blockDim.x + threadIdx.x;
    const size_t N = (size_t)T_ * B_ * KX;
    if (idx >= N) return;
    int k = (int)(idx % KX);
    int b = (int)((idx / KX) % B_);
    int t = (int)(idx / ((size_t)KX * B_));
    float v = 0.0f;
    if (k < C_)       v = xc[((size_t)b * T_ + t) * C_ + k];
    else if (k == C_) v = yf[(size_t)b * T_ + t];
    xpad[idx] = (bf16)v;
}

// Wt[n][k] (n in [0,2048), k in [0,608)): k<65 -> W[k][n]; 65<=k<96 -> 0; k>=96 -> W[k-31][n].
__global__ void pack_w_kernel(const float* __restrict__ W, bf16* __restrict__ wt) {
    size_t idx = (size_t)blockIdx.x * blockDim.x + threadIdx.x;
    const size_t N = (size_t)4 * H_ * KTOT;
    if (idx >= N) return;
    int k = (int)(idx % KTOT);
    int n = (int)(idx / KTOT);
    float v = 0.0f;
    if (k < C_ + 1)   v = W[(size_t)k * (4 * H_) + n];
    else if (k >= KX) v = W[(size_t)(k - (KX - C_ - 1)) * (4 * H_) + n]; // k-31
    wt[idx] = (bf16)v;
}

// zero h0 double buffer + grid-sync counter
__global__ void init_state_kernel(bf16* __restrict__ hbuf, unsigned* __restrict__ counter) {
    size_t idx = (size_t)blockIdx.x * blockDim.x + threadIdx.x;
    const size_t N = 2ull * B_ * H_;
    if (idx < N) hbuf[idx] = (bf16)0.0f;
    if (idx == 0) *counter = 0u;
}

// ---------------- persistent recurrent scan ----------------
// 64 persistent WGs: WG = (col group cg = wg&31, batch half bh = wg>>5).
// WG owns batch rows [32*bh, 32*bh+32) x hidden cols [16*cg, 16*cg+16), i.e. a
// 32x64 gate slice (four 16-wide i|f|g|o strips). Each of the 8 waves computes
// exactly one 16x16 WMMA tile (rt = wave&1, ct = wave>>1), K = 608 in 19 chunks.
// Wt fragments are register-resident for the whole scan (~152 VGPRs/wave); A
// fragments are bulk-loaded per step so ~one load latency hits the critical path.
__global__ void __launch_bounds__(NTH, 1)
lstm_scan_kernel(const bf16* __restrict__ Xpad,
                 const bf16* __restrict__ Wt,
                 bf16* __restrict__ hbuf,        // 2 x (B_ x H_), double buffered
                 const float* __restrict__ bias, // (4H)
                 float* __restrict__ out,        // hs | h_final | c_final
                 unsigned* __restrict__ counter) {
    __shared__ float sG[32][68];      // gate slice, padded to dodge bank conflicts
    __shared__ float sC[32][16];      // persistent cell state (our rows x cols)
    __shared__ float sBias[64];       // bias for our 64 gate cols (i|f|g|o strips)

    const int wg   = blockIdx.x;           // 0..63
    const int cg   = wg & 31;              // hidden-column group
    const int bh   = wg >> 5;              // batch half
    const int n0   = cg * 16;              // first hidden column owned
    const int row0 = bh * 32;              // first batch row owned
    const int tid  = threadIdx.x;
    const int lane = tid & 31;
    const int wave = tid >> 5;
    const int rt   = wave & 1;             // batch-tile row (0..1)
    const int ct   = wave >> 1;            // gate-tile col (0..3)
    const int r    = lane & 15;            // row/col within tile
    const int kk   = (lane >> 4) * 8;      // K sub-chunk base for this lane

    for (int i = tid; i < 32 * 16; i += NTH) sC[i >> 4][i & 15] = 0.0f;
    if (tid < 64) sBias[tid] = bias[(tid >> 4) * H_ + n0 + (tid & 15)];
    __syncthreads();

    // B-operand (Wt) fragments: loaded once, register-resident for the whole scan.
    const bf16* wtp = Wt + ((size_t)(ct * H_ + n0 + r)) * KTOT + kk;
    v16bf bf[NKC];
    #pragma unroll
    for (int kc = 0; kc < NKC; ++kc) bf[kc] = ld_frag(wtp + kc * 32);

    float* hs   = out;
    float* hfin = out + (size_t)B_ * T_ * H_;
    float* cfin = hfin + (size_t)B_ * H_;

    const int  arow  = row0 + rt * 16 + r;                 // this lane's A row (batch)
    const bf16* xbase = Xpad + (size_t)arow * KX + kk;

    for (int t = 0; t < T_; ++t) {
        const bf16* hcur  = hbuf + (size_t)(t & 1) * B_ * H_;
        bf16*       hnext = hbuf + (size_t)((t + 1) & 1) * B_ * H_;

        const bf16* xrow = xbase + (size_t)t * B_ * KX;
        const bf16* hrow = hcur + (size_t)arow * H_ + kk;

        // Bulk-load all A fragments for this step (loads issue back-to-back,
        // waits overlap with the WMMA chain below).
        v16bf af[NKC];
        #pragma unroll
        for (int kc = 0; kc < KX / 32; ++kc)
            af[kc] = ld_frag(xrow + kc * 32);
        #pragma unroll
        for (int kc = KX / 32; kc < NKC; ++kc)
            af[kc] = ld_frag(hrow + kc * 32 - KX);

        // Prefetch next step's x rows while the GEMM runs.
        if (t + 1 < T_) {
            const bf16* xn = xrow + (size_t)B_ * KX;
            __builtin_prefetch(xn, 0, 1);
            __builtin_prefetch(xn + 64, 0, 1);
        }

        v8f acc = {0.f,0.f,0.f,0.f,0.f,0.f,0.f,0.f};
        #pragma unroll
        for (int kc = 0; kc < NKC; ++kc)
            acc = __builtin_amdgcn_wmma_f32_16x16x32_bf16(
                      false, af[kc], false, bf[kc], (short)0, acc, false, false);

        // C/D layout (ISA 7.12.2): element v of lanes 0-15 -> row v, lanes 16-31 -> row 8+v
        #pragma unroll
        for (int v = 0; v < 8; ++v) {
            int row = rt * 16 + (lane >> 4) * 8 + v;     // local batch row 0..31
            sG[row][ct * 16 + r] = acc[v];
        }
        __syncthreads();

        // cell update: 32 batch rows x 16 hidden cols; 2 elems per thread
        #pragma unroll
        for (int j = 0; j < 2; ++j) {
            int idx = tid + j * NTH;
            int bl = idx >> 4, nl = idx & 15;
            float gi = sG[bl][nl]      + sBias[nl];
            float gf = sG[bl][16 + nl] + sBias[16 + nl];
            float gg = sG[bl][32 + nl] + sBias[32 + nl];
            float go = sG[bl][48 + nl] + sBias[48 + nl];
            float c  = sC[bl][nl];
            float cn = sigm_(gf) * c + sigm_(gi) * tanhf(gg);
            float hn = sigm_(go) * tanhf(cn);
            sC[bl][nl] = cn;
            int bb = row0 + bl;
            int ng = n0 + nl;
            // hs is write-only output: non-temporal so it never evicts Wt/h from L2
            __builtin_nontemporal_store(hn, &hs[((size_t)bb * T_ + t) * H_ + ng]);
            hnext[(size_t)bb * H_ + ng] = (bf16)hn;
            if (t == T_ - 1) {
                hfin[(size_t)bb * H_ + ng] = hn;
                cfin[(size_t)bb * H_ + ng] = cn;
            }
        }

        // grid-wide barrier: release h writes, arrive, spin, acquire
        __syncthreads();
        __threadfence();
        if (tid == 0) {
            __hip_atomic_fetch_add(counter, 1u, __ATOMIC_RELEASE, __HIP_MEMORY_SCOPE_AGENT);
            const unsigned target = (unsigned)NWG * (unsigned)(t + 1);
            while (__hip_atomic_load(counter, __ATOMIC_ACQUIRE, __HIP_MEMORY_SCOPE_AGENT) < target)
                __builtin_amdgcn_s_sleep(1);
        }
        __syncthreads();
        __threadfence();
    }
}

// ---------------- launcher ----------------
extern "C" void kernel_launch(void* const* d_in, const int* in_sizes, int n_in,
                              void* d_out, int out_size, void* d_ws, size_t ws_size,
                              hipStream_t stream) {
    const float* xc = (const float*)d_in[0];  // (B,T,C)
    const float* yf = (const float*)d_in[1];  // (B,T,1)
    const float* W  = (const float*)d_in[2];  // (577, 2048)
    const float* b  = (const float*)d_in[3];  // (2048,)
    float* out = (float*)d_out;

    char* ws = (char*)d_ws;
    const size_t XPAD_B = (size_t)T_ * B_ * KX * sizeof(bf16);     // 6,291,456
    const size_t WT_B   = (size_t)4 * H_ * KTOT * sizeof(bf16);    // 2,490,368
    const size_t HBUF_B = 2ull * B_ * H_ * sizeof(bf16);           // 131,072
    bf16*     xpad    = (bf16*)(ws);
    bf16*     wt      = (bf16*)(ws + XPAD_B);
    bf16*     hbuf    = (bf16*)(ws + XPAD_B + WT_B);
    unsigned* counter = (unsigned*)(ws + XPAD_B + WT_B + HBUF_B);

    {
        size_t n = (size_t)T_ * B_ * KX;
        pack_x_kernel<<<(unsigned)((n + NTH - 1) / NTH), NTH, 0, stream>>>(xc, yf, xpad);
    }
    {
        size_t n = (size_t)4 * H_ * KTOT;
        pack_w_kernel<<<(unsigned)((n + NTH - 1) / NTH), NTH, 0, stream>>>(W, wt);
    }
    {
        size_t n = 2ull * B_ * H_;
        init_state_kernel<<<(unsigned)((n + NTH - 1) / NTH), NTH, 0, stream>>>(hbuf, counter);
    }
    lstm_scan_kernel<<<NWG, NTH, 0, stream>>>(xpad, wt, hbuf, b, out, counter);
}
